// FQModel_3624952398776
// MI455X (gfx1250) — compile-verified, matrix operand order checked by reading, not run on hardware
//
#include <hip/hip_runtime.h>
#include <math.h>

typedef float v2f __attribute__((ext_vector_type(2)));
typedef float v8f __attribute__((ext_vector_type(8)));
typedef int   v8i __attribute__((ext_vector_type(8)));

#define NCH    3456      // number of chunks (rows)
#define CDIM   256       // embedding dim
#define KCODE  1024      // codebook entries
#define SPATN  216       // 6*6*6
#define NVOXN  13824     // 24^3
#define TINV   100.0f    // 1/TEMP
#define ZQ_ELEMS 884736  // 16*256*216

#define MT     108       // NCH/32 macro row tiles
#define NTRI   5886      // MT*(MT+1)/2 triangular macro tiles
#define PAIR_BLOCKS 736  // ceil(NTRI/8)
#define AFF_MACROS 3456  // 108 * 32 macro tiles (32x32 each)

__device__ __forceinline__ float waveSum(float v) {
#pragma unroll
  for (int off = 16; off > 0; off >>= 1) v += __shfl_xor(v, off, 32);
  return v;
}

// ---------------- 1) row-normalize z (gathered from z_e [B,C,6,6,6]) --------
__global__ void k_norm_z(const float* __restrict__ ze, float* __restrict__ zn) {
  const int wid = threadIdx.x >> 5, lane = threadIdx.x & 31;
  const int n = blockIdx.x * 8 + wid;              // 432 blocks * 8 = 3456
  const int b = n / SPATN, sidx = n - b * SPATN;
  const float* src = ze + (size_t)b * (CDIM * SPATN) + sidx;
  float v[8]; float ss = 0.f;
#pragma unroll
  for (int t = 0; t < 8; ++t) {
    const int c = lane + 32 * t;
    const float x = src[(size_t)c * SPATN];
    v[t] = x; ss += x * x;
  }
  ss = waveSum(ss);
  const float inv = 1.0f / fmaxf(sqrtf(ss), 1e-12f);
  float* dst = zn + (size_t)n * CDIM;
#pragma unroll
  for (int t = 0; t < 8; ++t) dst[lane + 32 * t] = v[t] * inv;
}

// ---------------- 2) row-normalize codebook ---------------------------------
__global__ void k_norm_cb(const float* __restrict__ cb, float* __restrict__ en) {
  const int wid = threadIdx.x >> 5, lane = threadIdx.x & 31;
  const int n = blockIdx.x * 8 + wid;              // 128 blocks * 8 = 1024
  const float* src = cb + (size_t)n * CDIM;
  float v[8]; float ss = 0.f;
#pragma unroll
  for (int t = 0; t < 8; ++t) { const float x = src[lane + 32 * t]; v[t] = x; ss += x * x; }
  ss = waveSum(ss);
  const float inv = 1.0f / fmaxf(sqrtf(ss), 1e-12f);
  float* dst = en + (size_t)n * CDIM;
#pragma unroll
  for (int t = 0; t < 8; ++t) dst[lane + 32 * t] = v[t] * inv;
}

// ---------------- 3) affinity = zn @ en^T, 2x2-blocked f32 WMMA -------------
// One wave computes a 32x32 macro tile: each loaded fragment feeds 2 WMMAs.
__global__ void k_gemm_aff(const float* __restrict__ zn, const float* __restrict__ en,
                           float* __restrict__ aff) {
  const int wid = threadIdx.x >> 5, lane = threadIdx.x & 31;
  const int tile = blockIdx.x * 8 + wid;           // 432 blocks * 8 = 3456 macros
  const int IT = tile >> 5;                        // 108 row macro tiles
  const int JT = tile & 31;                        // 32 col macro tiles
  const int m = lane & 15, g = lane >> 4, kb = 2 * g;
  const int i0 = IT * 32, j0 = JT * 32;
  const float* ApA = zn + (size_t)(i0 + m) * CDIM + kb;
  const float* ApB = zn + (size_t)(i0 + 16 + m) * CDIM + kb;
  const float* BpA = en + (size_t)(j0 + m) * CDIM + kb;
  const float* BpB = en + (size_t)(j0 + 16 + m) * CDIM + kb;
  v8f c00 = {}, c01 = {}, c10 = {}, c11 = {};
#pragma unroll 4
  for (int k = 0; k < CDIM; k += 4) {
    v2f aA = *(const v2f*)(ApA + k);
    v2f aB = *(const v2f*)(ApB + k);
    v2f bA = *(const v2f*)(BpA + k);
    v2f bB = *(const v2f*)(BpB + k);
    c00 = __builtin_amdgcn_wmma_f32_16x16x4_f32(false, aA, false, bA, (short)0, c00, false, false);
    c01 = __builtin_amdgcn_wmma_f32_16x16x4_f32(false, aA, false, bB, (short)0, c01, false, false);
    c10 = __builtin_amdgcn_wmma_f32_16x16x4_f32(false, aB, false, bA, (short)0, c10, false, false);
    c11 = __builtin_amdgcn_wmma_f32_16x16x4_f32(false, aB, false, bB, (short)0, c11, false, false);
  }
#pragma unroll
  for (int r = 0; r < 8; ++r) {
    const int row0 = i0 + r + 8 * g, row1 = row0 + 16;
    const int col0 = j0 + m,         col1 = col0 + 16;
    aff[(size_t)row0 * KCODE + col0] = c00[r];
    aff[(size_t)row0 * KCODE + col1] = c01[r];
    aff[(size_t)row1 * KCODE + col0] = c10[r];
    aff[(size_t)row1 * KCODE + col1] = c11[r];
  }
}

// ---------------- 4) per-row: argmax (first-occurrence), softmax stats ------
__global__ void k_row_stats(const float* __restrict__ aff, float* __restrict__ mrow,
                            float* __restrict__ rrow, float* __restrict__ plrow,
                            int* __restrict__ idx) {
  const int wid = threadIdx.x >> 5, lane = threadIdx.x & 31;
  const int n = blockIdx.x * 8 + wid;              // 432 * 8 = 3456
  const float* row = aff + (size_t)n * KCODE;
  float mv = -3.402823466e+38f; int mi = 0;
  for (int k = lane; k < KCODE; k += 32) {
    const float a = row[k];
    if (a > mv) { mv = a; mi = k; }                // strict > keeps first occurrence
  }
#pragma unroll
  for (int off = 16; off > 0; off >>= 1) {
    const float ov = __shfl_xor(mv, off, 32);
    const int   oi = __shfl_xor(mi, off, 32);
    if (ov > mv || (ov == mv && oi < mi)) { mv = ov; mi = oi; }
  }
  float S = 0.f, T1 = 0.f;                         // S = sum e^x, T1 = sum x*e^x
  for (int k = lane; k < KCODE; k += 32) {
    const float x = TINV * (row[k] - mv);
    const float e = expf(x);
    S += e; T1 += e * x;
  }
  S = waveSum(S); T1 = waveSum(T1);
  if (lane == 0) {
    const float lS = logf(S);
    mrow[n]  = mv;                                 // max affinity (codebook loss term)
    rrow[n]  = TINV * mv + lS;                     // log-normalizer for p = e^(100a - r)
    plrow[n] = T1 / S - lS;                        // sum_k p*log p
    idx[n]   = mi;
  }
}

// ---------------- 5) binary occupancy: argmax over 32 channels != 0 ---------
__global__ void k_binary(const float* __restrict__ in, unsigned char* __restrict__ pat) {
  const int f = blockIdx.x * blockDim.x + threadIdx.x;   // 864*256 = 221184 exact
  const int b = f / NVOXN, v = f - b * NVOXN;
  const float* base = in + (size_t)b * 32 * NVOXN + v;
  const float x0 = base[0];
  float best = -3.402823466e+38f;
#pragma unroll
  for (int c = 1; c < 32; ++c) best = fmaxf(best, base[(size_t)c * NVOXN]);
  pat[f] = (best > x0) ? (unsigned char)1 : (unsigned char)0;  // argmax!=0 <=> strict beat of ch0
}

// ---------------- 6) per-row popcounts of the 64-bit patterns ---------------
__global__ void k_pat_pop(const unsigned char* __restrict__ pat, int* __restrict__ s) {
  const int n = blockIdx.x * blockDim.x + threadIdx.x;
  if (n >= NCH) return;
  const unsigned char* p = pat + (size_t)n * 64;
  int acc = 0;
  for (int k = 0; k < 64; ++k) acc += p[k];
  s[n] = acc;
}

// ---------------- 7) fused symmetric pair kernel ----------------------------
// Triangular enumeration of 32x32 macro tiles (off-diagonal counted x2).
// Per macro: 2x2-blocked cos Gram (f32 WMMA, K=256) + 4 IU8 WMMAs (K=64
// == pattern length) for the binary Gram -> hamming-thresholded weights.
__global__ void k_pair(const float* __restrict__ zn, const unsigned char* __restrict__ pat,
                       const int* __restrict__ s, float* __restrict__ partials) {
  const int wid = threadIdx.x >> 5, lane = threadIdx.x & 31;
  const int tlin = blockIdx.x * 8 + wid;           // 736*8 = 5888 waves, 5886 used
  const bool valid = (tlin < NTRI);
  const int t = valid ? tlin : (NTRI - 1);         // clamp: EXEC stays all-ones
  // decode triangular (IT <= JT): base(IT) = IT*MT - IT*(IT-1)/2
  int IT = (int)((2.0 * MT + 1.0 - sqrt((2.0 * MT + 1.0) * (2.0 * MT + 1.0) - 8.0 * (double)t)) * 0.5);
  while ((IT + 1) * MT - ((IT + 1) * IT) / 2 <= t) ++IT;   // sqrt rounding fixup
  while (IT * MT - (IT * (IT - 1)) / 2 > t) --IT;
  const int JT = IT + (t - (IT * MT - (IT * (IT - 1)) / 2));
  const float fac = (IT == JT) ? 1.0f : 2.0f;

  const int m = lane & 15, g = lane >> 4, kb = 2 * g;
  const int i0 = IT * 32, j0 = JT * 32;

  // --- binary Gram: 4 x V_WMMA_I32_16X16X64_IU8 ---
  const unsigned char* pAa = pat + (size_t)(i0 + m) * 64 + 8 * g;
  const unsigned char* pAb = pat + (size_t)(i0 + 16 + m) * 64 + 8 * g;
  const unsigned char* pBa = pat + (size_t)(j0 + m) * 64 + 8 * g;
  const unsigned char* pBb = pat + (size_t)(j0 + 16 + m) * 64 + 8 * g;
  v8i avA, avB, bvA, bvB;
#pragma unroll
  for (int q = 0; q < 4; ++q) {
    const unsigned long long qa = *(const unsigned long long*)(pAa + 16 * q);
    const unsigned long long qb = *(const unsigned long long*)(pAb + 16 * q);
    const unsigned long long qc = *(const unsigned long long*)(pBa + 16 * q);
    const unsigned long long qd = *(const unsigned long long*)(pBb + 16 * q);
    avA[2 * q] = (int)(unsigned)(qa & 0xffffffffull); avA[2 * q + 1] = (int)(unsigned)(qa >> 32);
    avB[2 * q] = (int)(unsigned)(qb & 0xffffffffull); avB[2 * q + 1] = (int)(unsigned)(qb >> 32);
    bvA[2 * q] = (int)(unsigned)(qc & 0xffffffffull); bvA[2 * q + 1] = (int)(unsigned)(qc >> 32);
    bvB[2 * q] = (int)(unsigned)(qd & 0xffffffffull); bvB[2 * q + 1] = (int)(unsigned)(qd >> 32);
  }
  v8i z0 = {}, z1 = {}, z2 = {}, z3 = {};
  const v8i g00 = __builtin_amdgcn_wmma_i32_16x16x64_iu8(false, avA, false, bvA, z0, false, false);
  const v8i g01 = __builtin_amdgcn_wmma_i32_16x16x64_iu8(false, avA, false, bvB, z1, false, false);
  const v8i g10 = __builtin_amdgcn_wmma_i32_16x16x64_iu8(false, avB, false, bvA, z2, false, false);
  const v8i g11 = __builtin_amdgcn_wmma_i32_16x16x64_iu8(false, avB, false, bvB, z3, false, false);

  // --- cos Gram: 2x2-blocked f32 WMMA over K=256 (1 load per WMMA) ---
  const float* ApA = zn + (size_t)(i0 + m) * CDIM + kb;
  const float* ApB = zn + (size_t)(i0 + 16 + m) * CDIM + kb;
  const float* BpA = zn + (size_t)(j0 + m) * CDIM + kb;
  const float* BpB = zn + (size_t)(j0 + 16 + m) * CDIM + kb;
  v8f c00 = {}, c01 = {}, c10 = {}, c11 = {};
#pragma unroll 4
  for (int k = 0; k < CDIM; k += 4) {
    v2f aA = *(const v2f*)(ApA + k);
    v2f aB = *(const v2f*)(ApB + k);
    v2f bA = *(const v2f*)(BpA + k);
    v2f bB = *(const v2f*)(BpB + k);
    c00 = __builtin_amdgcn_wmma_f32_16x16x4_f32(false, aA, false, bA, (short)0, c00, false, false);
    c01 = __builtin_amdgcn_wmma_f32_16x16x4_f32(false, aA, false, bB, (short)0, c01, false, false);
    c10 = __builtin_amdgcn_wmma_f32_16x16x4_f32(false, aB, false, bA, (short)0, c10, false, false);
    c11 = __builtin_amdgcn_wmma_f32_16x16x4_f32(false, aB, false, bB, (short)0, c11, false, false);
  }

  // --- combine: ham = |Pi| + |Pj| - 2 Pi.Pj ; sim>0.92 <=> ham<=5 ---
  const int sjA = s[j0 + m], sjB = s[j0 + 16 + m];
  float aw = 0.f, ad = 0.f;
#pragma unroll
  for (int r = 0; r < 8; ++r) {
    const int iA = i0 + r + 8 * g, iB = iA + 16;
    const int jA = j0 + m,         jB = jA + 16;
    const int siA = s[iA], siB = s[iB];
    {
      const int ham = siA + sjA - 2 * g00[r];
      if (iA != jA && ham <= 5) { const float wgt = 1.0f - (float)ham * 0.015625f;
        aw += wgt; ad += wgt * (1.0f - c00[r]); }
    }
    {
      const int ham = siA + sjB - 2 * g01[r];
      if (iA != jB && ham <= 5) { const float wgt = 1.0f - (float)ham * 0.015625f;
        aw += wgt; ad += wgt * (1.0f - c01[r]); }
    }
    {
      const int ham = siB + sjA - 2 * g10[r];
      if (iB != jA && ham <= 5) { const float wgt = 1.0f - (float)ham * 0.015625f;
        aw += wgt; ad += wgt * (1.0f - c10[r]); }
    }
    {
      const int ham = siB + sjB - 2 * g11[r];
      if (iB != jB && ham <= 5) { const float wgt = 1.0f - (float)ham * 0.015625f;
        aw += wgt; ad += wgt * (1.0f - c11[r]); }
    }
  }
  aw *= fac; ad *= fac;
  if (!valid) { aw = 0.f; ad = 0.f; }
  aw = waveSum(aw); ad = waveSum(ad);
  __shared__ float sw[8], sd[8];
  if (lane == 0) { sw[wid] = aw; sd[wid] = ad; }
  __syncthreads();
  if (threadIdx.x == 0) {
    float tw = 0.f, td = 0.f;
#pragma unroll
    for (int q = 0; q < 8; ++q) { tw += sw[q]; td += sd[q]; }
    partials[2 * blockIdx.x]     = tw;
    partials[2 * blockIdx.x + 1] = td;
  }
}

// ---------------- 8) deterministic reduce over per-row stats ----------------
__global__ void k_reduce_rows(const float* __restrict__ m, const float* __restrict__ pl,
                              float* __restrict__ rowred) {
  __shared__ float s1[256], s2[256];
  float a1 = 0.f, a2 = 0.f;
  for (int n = threadIdx.x; n < NCH; n += 256) { a1 += m[n]; a2 += pl[n]; }
  s1[threadIdx.x] = a1; s2[threadIdx.x] = a2; __syncthreads();
  for (int off = 128; off > 0; off >>= 1) {
    if (threadIdx.x < off) { s1[threadIdx.x] += s1[threadIdx.x + off];
                             s2[threadIdx.x] += s2[threadIdx.x + off]; }
    __syncthreads();
  }
  if (threadIdx.x == 0) { rowred[0] = s1[0]; rowred[1] = s2[0]; }
}

// ---------------- 9) avg_probs column sums + entropy terms ------------------
__global__ void k_avg_probs(const float* __restrict__ aff, const float* __restrict__ r,
                            float* __restrict__ entterm) {
  const int k = blockIdx.x * blockDim.x + threadIdx.x;   // 4*256 = 1024
  float acc = 0.f;
  for (int n = 0; n < NCH; ++n)
    acc += expf(TINV * aff[(size_t)n * KCODE + k] - r[n]);   // p_nk in [0,1]
  const float ap = acc * (1.0f / (float)NCH);
  entterm[k] = ap * logf(ap + 1e-5f);
}

// ---------------- 10) deterministic reduce of pair partials -----------------
__global__ void k_reduce_pairs(const float* __restrict__ p, float* __restrict__ outp) {
  __shared__ float s1[256], s2[256];
  float a1 = 0.f, a2 = 0.f;
  for (int i = threadIdx.x; i < PAIR_BLOCKS; i += 256) { a1 += p[2 * i]; a2 += p[2 * i + 1]; }
  s1[threadIdx.x] = a1; s2[threadIdx.x] = a2; __syncthreads();
  for (int off = 128; off > 0; off >>= 1) {
    if (threadIdx.x < off) { s1[threadIdx.x] += s1[threadIdx.x + off];
                             s2[threadIdx.x] += s2[threadIdx.x + off]; }
    __syncthreads();
  }
  if (threadIdx.x == 0) { outp[0] = s1[0]; outp[1] = s2[0]; }
}

// ---------------- 11) write z_q (STE forward value = en[idx]) + idx ---------
__global__ void k_zq_idx(const float* __restrict__ en, const int* __restrict__ idx,
                         float* __restrict__ out) {
  const int t = blockIdx.x * blockDim.x + threadIdx.x;
  if (t < ZQ_ELEMS) {
    const int b = t / (CDIM * SPATN);
    const int rem = t - b * (CDIM * SPATN);
    const int c = rem / SPATN;
    const int sp = rem - c * SPATN;
    const int n = b * SPATN + sp;
    out[t] = en[(size_t)idx[n] * CDIM + c];
  } else if (t < ZQ_ELEMS + NCH) {
    const int n = t - ZQ_ELEMS;
    out[ZQ_ELEMS + 1 + n] = (float)idx[n];
  }
}

// ---------------- 12) assemble scalar loss ----------------------------------
__global__ void k_final(const float* __restrict__ entterm, const float* __restrict__ rowred,
                        const float* __restrict__ pairred, float* __restrict__ out) {
  __shared__ float sm[256];
  float acc = 0.f;
  for (int k = threadIdx.x; k < KCODE; k += 256) acc += entterm[k];
  sm[threadIdx.x] = acc; __syncthreads();
  for (int off = 128; off > 0; off >>= 1) {
    if (threadIdx.x < off) sm[threadIdx.x] += sm[threadIdx.x + off];
    __syncthreads();
  }
  if (threadIdx.x == 0) {
    const float Sae = sm[0];                       // sum ap*log(ap+1e-5)
    const float sum_m = rowred[0], sum_pl = rowred[1];
    // codebook + commit: (1+BETA)*mean((en[idx]-zn)^2) = 1.25*(2N - 2 sum max)/(N*C)
    const float cbc = 1.25f * (2.0f * (float)NCH - 2.0f * sum_m) / ((float)NCH * (float)CDIM);
    const float sample_ent = -sum_pl / (float)NCH;
    const float avg_ent = -Sae;
    const float ent = 0.1f * (sample_ent - avg_ent);
    const float ham = pairred[1] / (pairred[0] + 1e-8f);
    out[ZQ_ELEMS] = cbc + ent + ham;
  }
}

extern "C" void kernel_launch(void* const* d_in, const int* in_sizes, int n_in,
                              void* d_out, int out_size, void* d_ws, size_t ws_size,
                              hipStream_t stream) {
  const float* in_blocks = (const float*)d_in[0];   // [16,32,24,24,24]
  const float* z_e       = (const float*)d_in[1];   // [16,256,6,6,6]
  const float* codebook  = (const float*)d_in[2];   // [1024,256]
  float* out = (float*)d_out;                       // zq | loss | idx

  char* w = (char*)d_ws;
  float* zn   = (float*)w; w += (size_t)NCH * CDIM * 4;      // 3.54 MB
  float* en   = (float*)w; w += (size_t)KCODE * CDIM * 4;    // 1 MB
  float* aff  = (float*)w; w += (size_t)NCH * KCODE * 4;     // 14.2 MB
  float* mrow = (float*)w; w += NCH * 4;
  float* rrow = (float*)w; w += NCH * 4;
  float* plrow= (float*)w; w += NCH * 4;
  int*   idx  = (int*)w;   w += NCH * 4;
  unsigned char* pat = (unsigned char*)w; w += (size_t)NCH * 64;
  int*   srow = (int*)w;   w += NCH * 4;
  float* partials = (float*)w; w += (size_t)PAIR_BLOCKS * 2 * 4;
  float* rowred   = (float*)w; w += 16;
  float* pairred  = (float*)w; w += 16;
  float* entterm  = (float*)w; w += KCODE * 4;
  (void)in_sizes; (void)n_in; (void)out_size; (void)ws_size;

  k_norm_z     <<<dim3(432),  dim3(256), 0, stream>>>(z_e, zn);
  k_norm_cb    <<<dim3(128),  dim3(256), 0, stream>>>(codebook, en);
  k_gemm_aff   <<<dim3(432),  dim3(256), 0, stream>>>(zn, en, aff);
  k_row_stats  <<<dim3(432),  dim3(256), 0, stream>>>(aff, mrow, rrow, plrow, idx);
  k_binary     <<<dim3(864),  dim3(256), 0, stream>>>(in_blocks, pat);
  k_pat_pop    <<<dim3(14),   dim3(256), 0, stream>>>(pat, srow);
  k_pair       <<<dim3(PAIR_BLOCKS), dim3(256), 0, stream>>>(zn, pat, srow, partials);
  k_reduce_rows<<<dim3(1),    dim3(256), 0, stream>>>(mrow, plrow, rowred);
  k_avg_probs  <<<dim3(4),    dim3(256), 0, stream>>>(aff, rrow, entterm);
  k_reduce_pairs<<<dim3(1),   dim3(256), 0, stream>>>(partials, pairred);
  k_zq_idx     <<<dim3(3470), dim3(256), 0, stream>>>(en, idx, out);
  k_final      <<<dim3(1),    dim3(256), 0, stream>>>(entterm, rowred, pairred, out);
}
